// GraphAttention_83373905150278
// MI455X (gfx1250) — matrix operand order repaired and checked by reading, not compiled, not run on hardware
//
#include <hip/hip_runtime.h>
#include <hip/hip_bf16.h>

typedef __attribute__((ext_vector_type(16))) __bf16 v16bf;
typedef __attribute__((ext_vector_type(8)))  float  v8f;

__device__ __forceinline__ v8f wmma_bf16(v16bf a, v16bf b, v8f c) {
  return __builtin_amdgcn_wmma_f32_16x16x32_bf16(false, a, false, b, (short)0, c, false, false);
}

__device__ __forceinline__ unsigned encOrd(float f) {
  unsigned u = __float_as_uint(f);
  return (u & 0x80000000u) ? ~u : (u | 0x80000000u);
}
__device__ __forceinline__ float decOrd(unsigned v) {
  return __uint_as_float((v & 0x80000000u) ? (v & 0x7fffffffu) : ~v);
}

// per-lane async 16B copy: global -> LDS (tracked by ASYNCcnt)
__device__ __forceinline__ void async_cp16(const __bf16* gsrc, void* ldst) {
  unsigned lds = (unsigned)(uintptr_t)ldst;           // addr[31:0] = LDS offset
  asm volatile("global_load_async_to_lds_b128 %0, %1, off"
               :: "v"(lds), "v"(gsrc) : "memory");
}

// ---------------------------------------------------------------------------
// C[M x 256] = A_bf16[M x 256] @ W + bias, with WT[n][k] bf16 in global.
// Block: 256 threads (8 waves) -> 64 rows x 256 cols.
// Wave w: col-strip s=w&3 (64 cols), row-group g=w>>2 (32 rows) -> 8 WMMA acc.
// A and B staged in LDS in WMMA fragment-major order via async copies.
// ---------------------------------------------------------------------------
__global__ __launch_bounds__(256) void k_gemm256_bf16(
    const __bf16* __restrict__ A, const __bf16* __restrict__ WT,
    const float* __restrict__ bias, float* __restrict__ C, int M)
{
  __shared__ __attribute__((aligned(32))) __bf16 sB[65536]; // 128 KB: 16 nTiles x 8 ksteps x 32 lanes x 16 elems
  __shared__ __attribute__((aligned(32))) __bf16 sA[16384]; //  32 KB:  4 rTiles x 8 ksteps x 32 lanes x 16 elems

  const int tid  = threadIdx.x;
  const int lane = tid & 31;
  const int wave = tid >> 5;
  const int l16  = lane & 15;
  const int hi   = lane >> 4;
  const size_t blockRow = (size_t)blockIdx.x * 64;

  // ---- stage B (all of WT): 8192 chunks of 16B, fragment-major ----
  // frag f = nTile*8+kstep; lane region = 32B at sB[(f*32+l)*16]
  #pragma unroll
  for (int i = 0; i < 32; ++i) {
    int c = tid + i * 256;            // 0..8191
    int half  = c & 1;
    int l     = (c >> 1) & 31;
    int f     = c >> 6;               // 0..127
    int nTile = f >> 3, kstep = f & 7;
    int col   = nTile * 16 + (l & 15);
    int koff  = kstep * 32 + (l >> 4) * 16 + half * 8;
    async_cp16(WT + (size_t)col * 256 + koff, &sB[(f * 32 + l) * 16 + half * 8]);
  }
  // ---- stage A (64 rows of this block): 2048 chunks of 16B ----
  // frag f = rowTile*8+kstep; lane: h0 = A[row, kk+hi*16], h1 = A[row, kk+hi*16+8]
  #pragma unroll
  for (int i = 0; i < 8; ++i) {
    int c = tid + i * 256;            // 0..2047
    int half = c & 1;
    int l    = (c >> 1) & 31;
    int f    = c >> 6;                // 0..31
    int rowTile = f >> 3, kstep = f & 7;
    size_t row = blockRow + rowTile * 16 + (l & 15);
    int k = kstep * 32 + (l >> 4) * 16 + half * 8;
    async_cp16(A + row * 256 + k, &sA[(f * 32 + l) * 16 + half * 8]);
  }
  asm volatile("s_wait_asynccnt 0x0" ::: "memory");
  __syncthreads();

  const int s    = wave & 3;          // column strip (64 cols)
  const int gRow = wave >> 2;         // row group (32 rows)

  v8f acc[2][4] = {};
  #pragma unroll
  for (int kstep = 0; kstep < 8; ++kstep) {
    v16bf a0 = *(const v16bf*)&sA[(((gRow * 2 + 0) * 8 + kstep) * 32 + lane) * 16];
    v16bf a1 = *(const v16bf*)&sA[(((gRow * 2 + 1) * 8 + kstep) * 32 + lane) * 16];
    #pragma unroll
    for (int j = 0; j < 4; ++j) {
      v16bf b = *(const v16bf*)&sB[((((s * 4 + j) * 8) + kstep) * 32 + lane) * 16];
      acc[0][j] = wmma_bf16(a0, b, acc[0][j]);
      acc[1][j] = wmma_bf16(a1, b, acc[1][j]);
    }
  }

  #pragma unroll
  for (int t = 0; t < 2; ++t) {
    const size_t mBase = blockRow + (gRow * 2 + t) * 16 + hi * 8; // D row = r + 8*hi
    #pragma unroll
    for (int j = 0; j < 4; ++j) {
      const int col = s * 64 + j * 16 + l16;
      const float bv = bias[col];
      #pragma unroll
      for (int r = 0; r < 8; ++r)
        C[(mBase + r) * 256 + col] = acc[t][j][r] + bv;
    }
  }
}

// ---------------------------------------------------------------------------
__global__ void k_f32_to_bf16(const float* __restrict__ s, __bf16* __restrict__ d, int n) {
  int t = blockIdx.x * blockDim.x + threadIdx.x;
  if (t < n) d[t] = (__bf16)s[t];
}

// WT[n*256+k] = (bf16) W[k*256+n]   (256x256)
__global__ void k_w_transpose(const float* __restrict__ W, __bf16* __restrict__ WT) {
  int t = blockIdx.x * blockDim.x + threadIdx.x;
  if (t >= 65536) return;
  int n = t >> 8, k = t & 255;
  WT[t] = (__bf16)W[k * 256 + n];
}

__global__ void k_init_u32(unsigned* __restrict__ p, unsigned val, int n) {
  int t = blockIdx.x * blockDim.x + threadIdx.x;
  if (t < n) p[t] = val;
}

// eb[e*8+h] = be[h] + sum_j edge_attr[e*16+j] * We[j*8+h]
__global__ void k_edge_bias(const float* __restrict__ ea, const float* __restrict__ We,
                            const float* __restrict__ be, float* __restrict__ eb, int E) {
  int t = blockIdx.x * blockDim.x + threadIdx.x;
  if (t >= E * 8) return;
  int e = t >> 3, h = t & 7;
  float s = be[h];
  #pragma unroll
  for (int j = 0; j < 16; ++j) s += ea[e * 16 + j] * We[j * 8 + h];
  eb[t] = s;
}

// scores[e,b,h] = dot(Q[b,src,h,:], K[b,dst,h,:]) / sqrt(32) + eb[e,h]; atomic segment max
__global__ void k_scores(const float* __restrict__ Q, const float* __restrict__ K,
                         const int* __restrict__ ei, const float* __restrict__ eb,
                         float* __restrict__ sc, unsigned* __restrict__ smax, int E, int N) {
  int t = blockIdx.x * blockDim.x + threadIdx.x;
  if (t >= E * 16) return;
  int e = t >> 4, r = t & 15, b = r >> 3, h = r & 7;
  int src = ei[e], dst = ei[E + e];
  const float4* q4 = (const float4*)(Q + ((size_t)(b * N + src)) * 256 + h * 32);
  const float4* k4 = (const float4*)(K + ((size_t)(b * N + dst)) * 256 + h * 32);
  float s = 0.f;
  #pragma unroll
  for (int i = 0; i < 8; ++i) {
    float4 a = q4[i], c = k4[i];
    s += a.x * c.x + a.y * c.y + a.z * c.z + a.w * c.w;
  }
  s = s * 0.17677669529663687f + eb[e * 8 + h];   // 1/sqrt(32)
  sc[t] = s;
  atomicMax(&smax[src * 16 + r], encOrd(s));
}

// ex = exp(s - segmax[src]); atomic segment sum into denom
__global__ void k_expsum(const int* __restrict__ ei, float* __restrict__ sc,
                         const unsigned* __restrict__ smax, float* __restrict__ den,
                         int E, int N) {
  int t = blockIdx.x * blockDim.x + threadIdx.x;
  if (t >= E * 16) return;
  int e = t >> 4, r = t & 15;
  int src = ei[e];
  float ex = expf(sc[t] - decOrd(smax[src * 16 + r]));
  sc[t] = ex;
  atomicAdd(&den[src * 16 + r], ex);
}

// agg[b,src,h,k] += (ex/denom) * V[b,dst,h,k]; one thread per (e,b,h,k) -> coalesced
__global__ void k_aggregate(const int* __restrict__ ei, const float* __restrict__ sc,
                            const float* __restrict__ den, const float* __restrict__ V,
                            float* __restrict__ agg, int E, int N) {
  int t = blockIdx.x * blockDim.x + threadIdx.x;   // E*512 = 81.92M < 2^31
  int k = t & 31, h = (t >> 5) & 7, b = (t >> 8) & 1, e = t >> 9;
  if (e >= E) return;
  int src = ei[e], dst = ei[E + e];
  float w = sc[e * 16 + b * 8 + h] / den[src * 16 + b * 8 + h];
  float v = V[((size_t)(b * N + dst)) * 256 + h * 32 + k];
  atomicAdd(&agg[((size_t)(b * N + src)) * 256 + h * 32 + k], w * v);
}

// ---------------------------------------------------------------------------
extern "C" void kernel_launch(void* const* d_in, const int* in_sizes, int n_in,
                              void* d_out, int out_size, void* d_ws, size_t ws_size,
                              hipStream_t stream) {
  (void)in_sizes; (void)n_in; (void)out_size; (void)ws_size;
  const float* x  = (const float*)d_in[0];
  const int*   ei = (const int*)  d_in[1];
  const float* ea = (const float*)d_in[2];
  const float* Wq = (const float*)d_in[3];
  const float* bq = (const float*)d_in[4];
  const float* Wk = (const float*)d_in[5];
  const float* bk = (const float*)d_in[6];
  const float* Wv = (const float*)d_in[7];
  const float* bv = (const float*)d_in[8];
  const float* We = (const float*)d_in[9];
  const float* be = (const float*)d_in[10];
  const float* Wo = (const float*)d_in[11];
  const float* bo = (const float*)d_in[12];
  float* out = (float*)d_out;

  const int Bc = 2, Nn = 20000, Ec = 160000;
  const int M  = Bc * Nn;                 // 40000 GEMM rows (multiple of 64)
  const int nx = M * 256;

  char* ws = (char*)d_ws;
  size_t off = 0;
  auto take = [&](size_t bytes) -> char* {
    char* p = ws + off;
    off = (off + bytes + 255) & ~(size_t)255;
    return p;
  };
  float*    Qb   = (float*)   take((size_t)nx * 4);
  float*    Kb   = (float*)   take((size_t)nx * 4);
  float*    Vb   = (float*)   take((size_t)nx * 4);
  __bf16*   xbf  = (__bf16*)  take((size_t)nx * 2);
  __bf16*   WqT  = (__bf16*)  take(65536 * 2);
  __bf16*   WkT  = (__bf16*)  take(65536 * 2);
  __bf16*   WvT  = (__bf16*)  take(65536 * 2);
  __bf16*   WoT  = (__bf16*)  take(65536 * 2);
  float*    sc   = (float*)   take((size_t)Ec * 16 * 4);
  float*    eb   = (float*)   take((size_t)Ec * 8 * 4);
  unsigned* smax = (unsigned*)take((size_t)Nn * 16 * 4);
  float*    den  = (float*)   take((size_t)Nn * 16 * 4);
  float*    agg  = Qb;     // reuse Q region after score pass (stream-ordered)
  __bf16*   agbf = xbf;    // reuse x-bf16 region after QKV GEMMs

  // 1) input + weight conversion (f32 -> bf16, weights pre-transposed)
  k_f32_to_bf16<<<(nx + 255) / 256, 256, 0, stream>>>(x, xbf, nx);
  k_w_transpose<<<256, 256, 0, stream>>>(Wq, WqT);
  k_w_transpose<<<256, 256, 0, stream>>>(Wk, WkT);
  k_w_transpose<<<256, 256, 0, stream>>>(Wv, WvT);
  k_w_transpose<<<256, 256, 0, stream>>>(Wo, WoT);

  // 2) Q/K/V projections via WMMA (64-row blocks, LDS-staged operands)
  k_gemm256_bf16<<<M / 64, 256, 0, stream>>>(xbf, WqT, bq, Qb, M);
  k_gemm256_bf16<<<M / 64, 256, 0, stream>>>(xbf, WkT, bk, Kb, M);
  k_gemm256_bf16<<<M / 64, 256, 0, stream>>>(xbf, WvT, bv, Vb, M);

  // 3) edge bias (tiny GEMV)
  k_edge_bias<<<(Ec * 8 + 255) / 256, 256, 0, stream>>>(ea, We, be, eb, Ec);

  // 4) segment softmax state init (enc(-inf) = 0x007FFFFF)
  k_init_u32<<<(Nn * 16 + 255) / 256, 256, 0, stream>>>(smax, 0x007FFFFFu, Nn * 16);
  (void)hipMemsetAsync(den, 0, (size_t)Nn * 16 * 4, stream);

  // 5) scores + segment max (L2-resident gathers)
  k_scores<<<(Ec * 16 + 255) / 256, 256, 0, stream>>>(Qb, Kb, ei, eb, sc, smax, Ec, Nn);

  // 6) zero agg (aliases Q; safe now that scores are done)
  (void)hipMemsetAsync(agg, 0, (size_t)nx * 4, stream);

  // 7) exp + segment sum
  k_expsum<<<(Ec * 16 + 255) / 256, 256, 0, stream>>>(ei, sc, smax, den, Ec, Nn);

  // 8) weighted V aggregation (coalesced f32 atomics into L2)
  k_aggregate<<<(Ec * 512) / 256, 256, 0, stream>>>(ei, sc, den, Vb, agg, Ec, Nn);

  // 9) output projection via WMMA
  k_f32_to_bf16<<<(nx + 255) / 256, 256, 0, stream>>>(agg, agbf, nx);
  k_gemm256_bf16<<<M / 64, 256, 0, stream>>>(agbf, WoT, bo, out, M);
}
